// AttentionCritic_13683765805448
// MI455X (gfx1250) — matrix operand order repaired and checked by reading, not compiled, not run on hardware
//
#include <hip/hip_runtime.h>
#include <hip/hip_bf16.h>

typedef _Float16 h16;
typedef __attribute__((ext_vector_type(16))) _Float16 v16h;
typedef __attribute__((ext_vector_type(8)))  _Float16 v8h;
typedef __attribute__((ext_vector_type(8)))  float    v8f;

#define NA       512
#define KOBS     4
#define NIMG     2048        // NA*KOBS
#define CIN      3
#define HWDIM    21
#define NPIX     441
#define NPIX_PAD 448
#define HIDC     64
#define C2       16
#define KP       7056        // 16*441
#define KPAD     7168
#define KSTEPS   224         // KPAD/32
#define ACTD     5
#define OBSR     10

// ---- WMMA fragment index maps (CDNA5 ISA 7.12.2, wave32) ----
// 16-bit A 16x32: lanes0-15 rows M=0..15 hold K {0-7,16-23}; lanes16-31 hold K {8-15,24-31}
__device__ __forceinline__ int khalfA(int lane, int e) {
  return (lane < 16) ? (e < 8 ? e : e + 8) : (e < 8 ? e + 8 : e + 16);
}
// 16-bit B 32x16: lanes0-15 col N=0..15 hold K {0-7,16-23}; lanes16-31 hold K {8-15,24-31}
__device__ __forceinline__ int kmapB(int lane, int e) {
  return (lane < 16) ? (e < 8 ? e : e + 8) : (e < 8 ? e + 8 : e + 16);
}

__device__ __forceinline__ v8f wmma16(v16h a, v16h b, v8f c) {
  return __builtin_amdgcn_wmma_f32_16x16x32_f16(false, a, false, b, (short)0, c, false, false);
}

// combine two 16B chunks into a v16h fragment register set
__device__ __forceinline__ v16h join8(v8h lo, v8h hi) {
  return __builtin_shufflevector(lo, hi, 0,1,2,3,4,5,6,7,8,9,10,11,12,13,14,15);
}

// =====================================================================
// Kernel 0: repack weights into fragment-native f16 blobs (32B/lane chunks)
//   Bc1 : conv1  [ntile(4)][lane(32)][e(16)]  B[k][n]=conv1_w[n][k], k<27
//   Bc2 : conv2  [s(2)][lane][e]              B[k][n]=conv2_w[n][k], k<64
//   Bobs: linear [s(224)][lane][e]            B[k][n]=obs_w[n][k],   k<7056
// =====================================================================
__global__ void prep_kernel(const float* __restrict__ conv1_w,
                            const float* __restrict__ conv2_w,
                            const float* __restrict__ obs_w,
                            h16* __restrict__ Bc1, h16* __restrict__ Bc2,
                            h16* __restrict__ Bobs) {
  int i = blockIdx.x * 256 + threadIdx.x;
  if (i < 2048) {                       // Bc1: 4*32*16
    int e = i & 15, l = (i >> 4) & 31, nt = i >> 9;
    int n = nt * 16 + (l & 15);
    int k = kmapB(l, e);
    Bc1[i] = (k < 27) ? (h16)conv1_w[n * 27 + k] : (h16)0.0f;
  } else if (i < 3072) {                // Bc2: 2*32*16
    int j = i - 2048;
    int e = j & 15, l = (j >> 4) & 31, s = j >> 9;
    int n = l & 15;
    int k = s * 32 + kmapB(l, e);
    Bc2[j] = (h16)conv2_w[n * 64 + k];
  } else if (i < 117760) {              // Bobs: 224*32*16
    int j = i - 3072;
    int e = j & 15, l = (j >> 4) & 31, s = j >> 9;
    int n = l & 15;
    int k = s * 32 + kmapB(l, e);
    Bobs[j] = (k < KP) ? (h16)obs_w[n * KP + k] : (h16)0.0f;
  }
}

// =====================================================================
// Kernel 1: fused conv1(3x3,3->64)+ReLU and conv2(1x1,64->16)+ReLU
// one block per image; WMMA implicit GEMM, intermediates LDS-resident.
// conv1: each wave builds one im2col A fragment and reuses it for all
// 4 ntile WMMAs (B fragments are loop-invariant, preloaded).
// x2 out layout: [image][c*441 + p] f16, row stride KPAD (pad zeroed)
// =====================================================================
__global__ __launch_bounds__(256)
void conv_fused_kernel(const float* __restrict__ obs,
                       const float* __restrict__ conv1_b,
                       const float* __restrict__ conv2_b,
                       const h16* __restrict__ Bc1,
                       const h16* __restrict__ Bc2,
                       h16* __restrict__ x2) {
  __shared__ h16 imgp[CIN * 23 * 23];        // zero-padded image, f16
  __shared__ h16 x1[NPIX_PAD * HIDC];        // conv1 output (pixels x 64ch)

  const int tid  = threadIdx.x;
  const int wave = tid >> 5;
  const int lane = tid & 31;
  const int img  = blockIdx.x;

  // stage image with SAME zero-pad
  for (int i = tid; i < CIN * 23 * 23; i += 256) imgp[i] = (h16)0.0f;
  __syncthreads();
  const float* src = obs + (size_t)img * (CIN * NPIX);
  for (int i = tid; i < CIN * NPIX; i += 256) {
    int c = i / NPIX, p = i - c * NPIX;
    int py = p / HWDIM, px = p - py * HWDIM;
    imgp[c * 529 + (py + 1) * 23 + (px + 1)] = (h16)src[i];
  }
  __syncthreads();

  // ---- phase 1: conv1 as GEMM  M=448 pixels, N=64 ch, K=32 (27 used) ----
  // preload the 4 B fragments (loop-invariant per wave)
  v16h b0 = *(const v16h*)(Bc1 + (0 * 32 + lane) * 16);
  v16h b1 = *(const v16h*)(Bc1 + (1 * 32 + lane) * 16);
  v16h b2 = *(const v16h*)(Bc1 + (2 * 32 + lane) * 16);
  v16h b3 = *(const v16h*)(Bc1 + (3 * 32 + lane) * 16);
  const float bias1_0 = conv1_b[ 0 + (lane & 15)];
  const float bias1_1 = conv1_b[16 + (lane & 15)];
  const float bias1_2 = conv1_b[32 + (lane & 15)];
  const float bias1_3 = conv1_b[48 + (lane & 15)];
  const int mbase = (lane < 16) ? 0 : 8;

  for (int mtile = wave; mtile < 28; mtile += 8) {
    // build im2col A fragment ONCE per mtile
    int p = mtile * 16 + (lane & 15);
    if (p > NPIX - 1) p = NPIX - 1;          // clamp pad rows (discarded)
    int py = p / HWDIM, px = p - py * HWDIM;
    v16h a;
#pragma unroll
    for (int e = 0; e < 16; ++e) {
      int k = khalfA(lane, e);
      h16 v = (h16)0.0f;
      if (k < 27) {
        int c = k / 9, t = k - 9 * c;
        int ky = t / 3, kx = t - 3 * ky;
        v = imgp[c * 529 + (py + ky) * 23 + (px + kx)];
      }
      a[e] = v;
    }
    // 4 WMMAs against the 4 resident B fragments
#pragma unroll
    for (int nt = 0; nt < 4; ++nt) {
      v16h b = (nt == 0) ? b0 : (nt == 1) ? b1 : (nt == 2) ? b2 : b3;
      float bias = (nt == 0) ? bias1_0 : (nt == 1) ? bias1_1
                 : (nt == 2) ? bias1_2 : bias1_3;
      v8f d = {};
      d = wmma16(a, b, d);
#pragma unroll
      for (int e = 0; e < 8; ++e) {
        int pix = mtile * 16 + mbase + e;
        float r = d[e] + bias;
        r = r > 0.0f ? r : 0.0f;
        x1[pix * HIDC + nt * 16 + (lane & 15)] = (h16)r;
      }
    }
  }
  __syncthreads();

  // ---- phase 2: conv2 1x1 as GEMM  M=448, N=16, K=64 (2 k-steps) ----
  h16* out = x2 + (size_t)img * KPAD;
  for (int mtile = wave; mtile < 28; mtile += 8) {
    v8f acc = {};
#pragma unroll
    for (int s = 0; s < 2; ++s) {
      v16h b = *(const v16h*)(Bc2 + (s * 32 + lane) * 16);
      int row  = mtile * 16 + (lane & 15);
      int base = s * 32 + ((lane < 16) ? 0 : 8);
      v8h lo = *(const v8h*)(x1 + row * HIDC + base);
      v8h hi = *(const v8h*)(x1 + row * HIDC + base + 16);
      acc = wmma16(join8(lo, hi), b, acc);
    }
    float bias = conv2_b[lane & 15];
#pragma unroll
    for (int e = 0; e < 8; ++e) {
      int pix = mtile * 16 + mbase + e;
      if (pix < NPIX) {
        float r = acc[e] + bias;
        r = r > 0.0f ? r : 0.0f;
        out[(lane & 15) * NPIX + pix] = (h16)r;   // flatten order c*441+p
      }
    }
  }
  if (tid < KPAD - KP) out[KP + tid] = (h16)0.0f; // zero K pad
}

// =====================================================================
// Kernel 2: obs linear  [2048 x 7168] x [7168 x 16] -> o[2048][16] f32
// split-K over 8 waves (28 k-steps each), LDS reduction. No bias here.
// =====================================================================
__global__ __launch_bounds__(256)
void gemm_obs_kernel(const h16* __restrict__ x2,
                     const h16* __restrict__ Bobs,
                     float* __restrict__ o) {
  __shared__ float partial[8 * 256];
  const int tid  = threadIdx.x;
  const int wave = tid >> 5;
  const int lane = tid & 31;
  const int mtile = blockIdx.x;

  v8f acc = {};
  const int row = mtile * 16 + (lane & 15);
  const h16* arow = x2 + (size_t)row * KPAD + ((lane < 16) ? 0 : 8);
  for (int s = wave; s < KSTEPS; s += 8) {
    v16h b = *(const v16h*)(Bobs + (s * 32 + lane) * 16);
    v8h lo = *(const v8h*)(arow + s * 32);
    v8h hi = *(const v8h*)(arow + s * 32 + 16);
    acc = wmma16(join8(lo, hi), b, acc);
  }
  int mbase = (lane < 16) ? 0 : 8;
#pragma unroll
  for (int e = 0; e < 8; ++e)
    partial[wave * 256 + (mbase + e) * 16 + (lane & 15)] = acc[e];
  __syncthreads();
  float s = 0.0f;
#pragma unroll
  for (int w = 0; w < 8; ++w) s += partial[w * 256 + tid];
  int m = tid >> 4, n = tid & 15;
  o[(size_t)(mtile * 16 + m) * 16 + n] = s;
}

// =====================================================================
// Kernel 3: mean over K_obs (+obs_b, bias commutes with mean) and
// action encoder (one-hot @ act_w.T == column gather) -> cfeat[512][32]
// =====================================================================
__global__ void feat_kernel(const float* __restrict__ o,
                            const float* __restrict__ obs_b,
                            const int* __restrict__ action,
                            const float* __restrict__ act_w,
                            const float* __restrict__ act_b,
                            float* __restrict__ cfeat) {
  int n = blockIdx.x, t = threadIdx.x;
  if (t < 16) {
    float s = 0.0f;
#pragma unroll
    for (int k = 0; k < KOBS; ++k) s += o[(size_t)(n * KOBS + k) * 16 + t];
    cfeat[n * 32 + t] = 0.25f * s + obs_b[t];
  } else {
    int j = t - 16;
    int a = action[n];
    cfeat[n * 32 + t] = act_w[j * ACTD + a] + act_b[j];
  }
}

// =====================================================================
// Kernel 4: proximity mask, masked feature sum, dueling head.
// one wave per agent; lane t owns feature dim t.
// =====================================================================
__global__ void head_kernel(const float* __restrict__ cfeat,
                            const int* __restrict__ state,
                            const float* __restrict__ val_w,
                            const float* __restrict__ val_b,
                            const float* __restrict__ adv_w,
                            const float* __restrict__ adv_b,
                            float* __restrict__ out) {
  __shared__ float lat[32];
  __shared__ float hd[6];
  const int i = blockIdx.x, t = threadIdx.x;
  const int sx = state[2 * i], sy = state[2 * i + 1];

  float acc = 0.0f;
  for (int j = 0; j < NA; ++j) {
    bool obs;
    if (j == i) obs = true;
    else if (j > i) {
      int dx = sx - state[2 * j]; if (dx < 0) dx = -dx;
      int dy = sy - state[2 * j + 1]; if (dy < 0) dy = -dy;
      obs = (dx <= OBSR) && (dy <= OBSR);
    } else obs = false;
    if (obs) acc += cfeat[j * 32 + t];
  }
  lat[t] = acc;
  __syncthreads();
  if (t < 6) {
    const float* w = (t == 0) ? val_w : (adv_w + (t - 1) * 32);
    float b = (t == 0) ? val_b[0] : adv_b[t - 1];
    float s = b;
#pragma unroll
    for (int u = 0; u < 32; ++u) s += lat[u] * w[u];
    hd[t] = s;
  }
  __syncthreads();
  if (t == 0) {
    float V = hd[0];
    float mA = 0.2f * (hd[1] + hd[2] + hd[3] + hd[4] + hd[5]);
#pragma unroll
    for (int d = 0; d < ACTD; ++d) out[i * ACTD + d] = V + hd[1 + d] - mA;
  }
}

extern "C" void kernel_launch(void* const* d_in, const int* in_sizes, int n_in,
                              void* d_out, int out_size, void* d_ws, size_t ws_size,
                              hipStream_t stream) {
  const float* obs     = (const float*)d_in[0];
  const int*   action  = (const int*)d_in[1];
  const int*   state   = (const int*)d_in[2];
  const float* conv1_w = (const float*)d_in[3];
  const float* conv1_b = (const float*)d_in[4];
  const float* conv2_w = (const float*)d_in[5];
  const float* conv2_b = (const float*)d_in[6];
  const float* obs_w   = (const float*)d_in[7];
  const float* obs_b   = (const float*)d_in[8];
  const float* act_w   = (const float*)d_in[9];
  const float* act_b   = (const float*)d_in[10];
  const float* val_w   = (const float*)d_in[11];
  const float* val_b   = (const float*)d_in[12];
  const float* adv_w   = (const float*)d_in[13];
  const float* adv_b   = (const float*)d_in[14];
  float* out = (float*)d_out;

  char* ws = (char*)d_ws;
  h16*   Bc1   = (h16*)(ws + 0);                      // 4096 B
  h16*   Bc2   = (h16*)(ws + 4096);                   // 2048 B
  h16*   Bobs  = (h16*)(ws + 8192);                   // 229376 B
  h16*   x2    = (h16*)(ws + 237568);                 // 2048*7168*2 B
  float* o     = (float*)(ws + 237568 + (size_t)NIMG * KPAD * 2);
  float* cfeat = (float*)((char*)o + (size_t)NIMG * 16 * 4);

  prep_kernel<<<(117760 + 255) / 256, 256, 0, stream>>>(conv1_w, conv2_w, obs_w,
                                                        Bc1, Bc2, Bobs);
  conv_fused_kernel<<<NIMG, 256, 0, stream>>>(obs, conv1_b, conv2_b, Bc1, Bc2, x2);
  gemm_obs_kernel<<<NIMG / 16, 256, 0, stream>>>(x2, Bobs, o);
  feat_kernel<<<NA, 32, 0, stream>>>(o, obs_b, action, act_w, act_b, cfeat);
  head_kernel<<<NA, 32, 0, stream>>>(cfeat, state, val_w, val_b, adv_w, adv_b, out);
}
